// CasualSelfAttention_72404558676420
// MI455X (gfx1250) — matrix operand order repaired and checked by reading, not compiled, not run on hardware
//
#include <hip/hip_runtime.h>
#include <hip/hip_bf16.h>

// ---------------------------------------------------------------------------
// Causal self-attention (GPT-2 style), B=8 T=1024 C=768 H=12 D=64
// bf16 WMMA pipeline for MI455X (gfx1250, wave32).
//   - GEMMs: 128x128 block tile, 32x64 wave tile (8 accумulators/wave),
//     weight tile staged to LDS via GLOBAL_LOAD_ASYNC_TO_LDS_B128 (ASYNCcnt).
//   - Attention: flash-style, online softmax in f32 C-fragments,
//     P transposed through wave-private LDS tile.
// ---------------------------------------------------------------------------

#define B_ 8
#define T_ 1024
#define C_ 768
#define H_ 12
#define D_ 64
#define M_ (B_ * T_)   // 8192

typedef __attribute__((ext_vector_type(16))) __bf16 v16bf;
typedef __attribute__((ext_vector_type(8)))  float  v8f;

union AF { v16bf v; unsigned int d[8]; };
union CF { v8f   v; float        c[8]; };

__device__ __forceinline__ unsigned short f2bf(float f) {
    unsigned int u = __float_as_uint(f);
    u += 0x7fffu + ((u >> 16) & 1u);            // round to nearest even
    return (unsigned short)(u >> 16);
}

// pair index (units of 2 bf16) for VGPR j of a 16-bit 16x32 A / 32x16 B fragment
__device__ __forceinline__ int pair_base(int j, int hi) {
    return ((j & 4) ? (8 + (j & 3)) : j) + (hi << 2);
}

// direct global -> LDS async copy of 16 bytes (per lane), ASYNCcnt-tracked
__device__ __forceinline__ void async_ld_b128(unsigned int lds_byte_addr,
                                              const void* gaddr) {
    asm volatile("global_load_async_to_lds_b128 %0, %1, off"
                 :: "v"(lds_byte_addr), "v"(gaddr) : "memory");
}
__device__ __forceinline__ void wait_async0() {
    asm volatile("s_wait_asynccnt 0x0" ::: "memory");
}

// ---------------------------------------------------------------------------
// Conversion helpers
// ---------------------------------------------------------------------------
__global__ void csa_cvt_f32_bf16(const float* __restrict__ s,
                                 unsigned short* __restrict__ d, int n) {
    int i = blockIdx.x * blockDim.x + threadIdx.x;
    if (i < n) d[i] = f2bf(s[i]);
}

// src is [K,N] row-major, dst is [N,K] row-major (transposed), bf16
__global__ void csa_tr_f32_bf16(const float* __restrict__ s,
                                unsigned short* __restrict__ d, int K, int N) {
    int i = blockIdx.x * blockDim.x + threadIdx.x;
    if (i < N * K) {
        int n = i / K, k = i - n * K;
        d[i] = f2bf(s[(size_t)k * N + n]);
    }
}

// ---------------------------------------------------------------------------
// Tiled bf16 WMMA GEMM: Cout = A[M,K] * W[K,N] + bias, with W given as
// Wt[N,K] (pre-transposed).  Block = 8 waves (4 along M x 2 along N),
// block tile 128(M) x 128(N), wave tile 32(M) x 64(N) = 8 C fragments.
// mode 0: scatter into q [B,H,T,D], k [B,H,T,D], v-transposed [B,H,D,T] (bf16)
// mode 1: plain f32 row-major output [M, 768]
// ---------------------------------------------------------------------------
__global__ __launch_bounds__(256) void csa_gemm_wmma(
    const unsigned short* __restrict__ A,    // [M, K] bf16
    const unsigned short* __restrict__ Wt,   // [N, K] bf16
    const float* __restrict__ bias,          // [N]
    int K, int mode,
    unsigned short* __restrict__ qb,
    unsigned short* __restrict__ kb,
    unsigned short* __restrict__ vtb,
    float* __restrict__ outf)
{
    __shared__ __align__(16) unsigned short ldsB[128 * 32];  // 8 KB tile of Wt

    const int tid = threadIdx.x;
    const int w   = tid >> 5;
    const int wm  = w & 3;                  // M sub-tile 0..3
    const int wn  = w >> 2;                 // N half 0..1
    const int l   = tid & 31;
    const int hi  = l >> 4;
    const int ln  = l & 15;
    const int m0  = blockIdx.x * 128 + wm * 32;
    const int bn0 = blockIdx.y * 128;       // block's first output column
    const int n0  = bn0 + wn * 64;          // wave's first output column

    CF acc[2][4];
#pragma unroll
    for (int mi = 0; mi < 2; ++mi)
#pragma unroll
        for (int nt = 0; nt < 4; ++nt)
#pragma unroll
            for (int r = 0; r < 8; ++r) acc[mi][nt].c[r] = 0.0f;

    const unsigned short* Arow0 = A + (size_t)(m0 + ln) * K;
    const unsigned short* Arow1 = A + (size_t)(m0 + 16 + ln) * K;

    // LDS staging task: 128 rows x 2 sixteen-element halves = 256 threads
    const int srow  = tid >> 1;             // 0..127
    const int scol  = (tid & 1) * 16;       // 0 or 16 (ushort)
    const unsigned short* gstage0 = Wt + (size_t)(bn0 + srow) * K + scol;
    const unsigned int lds_stage =
        (unsigned int)(size_t)(ldsB + srow * 32 + scol);

    for (int kc = 0; kc < K; kc += 32) {
        // async stage Wt[bn0..bn0+127, kc..kc+31] straight into LDS
        async_ld_b128(lds_stage, gstage0 + kc);
        async_ld_b128(lds_stage + 16, gstage0 + kc + 8);
        if (kc + 32 < K) {                  // prefetch next A chunk
            __builtin_prefetch(Arow0 + kc + 32, 0, 0);
            __builtin_prefetch(Arow1 + kc + 32, 0, 0);
        }
        wait_async0();
        __syncthreads();

        AF a0, a1;
        {
            const unsigned int* ap0 = (const unsigned int*)(Arow0 + kc);
            const unsigned int* ap1 = (const unsigned int*)(Arow1 + kc);
#pragma unroll
            for (int j = 0; j < 8; ++j) {
                const int p = pair_base(j, hi);
                a0.d[j] = ap0[p];
                a1.d[j] = ap1[p];
            }
        }
        const unsigned int* lb = (const unsigned int*)ldsB;
#pragma unroll
        for (int nt = 0; nt < 4; ++nt) {
            AF b;
            const unsigned int* bp = lb + (wn * 64 + nt * 16 + ln) * 16;
#pragma unroll
            for (int j = 0; j < 8; ++j) b.d[j] = bp[pair_base(j, hi)];
            acc[0][nt].v = __builtin_amdgcn_wmma_f32_16x16x32_bf16(
                false, a0.v, false, b.v, (short)0, acc[0][nt].v, false, false);
            acc[1][nt].v = __builtin_amdgcn_wmma_f32_16x16x32_bf16(
                false, a1.v, false, b.v, (short)0, acc[1][nt].v, false, false);
        }
        __syncthreads();
    }

    if (mode == 0) {
#pragma unroll
        for (int nt = 0; nt < 4; ++nt) {
            const int col = n0 + nt * 16 + ln;     // 0..2303
            const int g   = col / C_;              // 0=q 1=k 2=v
            const int c   = col - g * C_;
            const int hh  = c >> 6;
            const int dd  = c & 63;
            const float bv = bias[col];
#pragma unroll
            for (int mi = 0; mi < 2; ++mi)
#pragma unroll
                for (int r = 0; r < 8; ++r) {
                    const int m  = m0 + mi * 16 + r + hi * 8;
                    const int bb = m >> 10;
                    const int t  = m & 1023;
                    const unsigned short val = f2bf(acc[mi][nt].c[r] + bv);
                    const size_t bh = (size_t)(bb * H_ + hh);
                    if (g == 0)      qb [(bh * T_ + t) * D_ + dd] = val;
                    else if (g == 1) kb [(bh * T_ + t) * D_ + dd] = val;
                    else             vtb[(bh * D_ + dd) * T_ + t] = val;  // V^T
                }
        }
    } else {
#pragma unroll
        for (int nt = 0; nt < 4; ++nt) {
            const int col = n0 + nt * 16 + ln;
            const float bv = bias[col];
#pragma unroll
            for (int mi = 0; mi < 2; ++mi)
#pragma unroll
                for (int r = 0; r < 8; ++r) {
                    const int m = m0 + mi * 16 + r + hi * 8;
                    outf[(size_t)m * C_ + col] = acc[mi][nt].c[r] + bv;
                }
        }
    }
}

// ---------------------------------------------------------------------------
// Flash-style causal attention. One wave per 16-row q tile; 8 waves/block
// cover 128 q rows of one (b,h). Online softmax in f32 C-fragments.
// ---------------------------------------------------------------------------
__global__ __launch_bounds__(256) void csa_attn_wmma(
    const unsigned short* __restrict__ qb,   // [B,H,T,D] bf16
    const unsigned short* __restrict__ kb,   // [B,H,T,D] bf16
    const unsigned short* __restrict__ vtb,  // [B,H,D,T] bf16
    unsigned short* __restrict__ ao)         // [B,T,C]  bf16
{
    __shared__ __align__(16) unsigned short ldsP[8 * 16 * 32];  // 8 KB, per-wave P tile

    const int tid = threadIdx.x;
    const int w   = tid >> 5;
    const int l   = tid & 31;
    const int hi  = l >> 4;
    const int ln  = l & 15;
    const int bh  = blockIdx.x;
    const int bb  = bh / H_;
    const int hh  = bh - bb * H_;
    const int q0  = blockIdx.y * 128 + w * 16;

    const unsigned short* Q  = qb  + (size_t)bh * T_ * D_;
    const unsigned short* Kp = kb  + (size_t)bh * T_ * D_;
    const unsigned short* Vt = vtb + (size_t)bh * D_ * T_;

    // Q A-fragments for k-dim (D=64) chunks 0..31 and 32..63 — loop invariant
    AF aq[2];
    {
        const unsigned int* qp = (const unsigned int*)(Q + (size_t)(q0 + ln) * D_);
#pragma unroll
        for (int dk = 0; dk < 2; ++dk)
#pragma unroll
            for (int j = 0; j < 8; ++j)
                aq[dk].d[j] = qp[dk * 16 + pair_base(j, hi)];
    }

    float mrun[8], lsum[8];
#pragma unroll
    for (int r = 0; r < 8; ++r) { mrun[r] = -1e30f; lsum[r] = 0.0f; }
    CF o[4];
#pragma unroll
    for (int nt = 0; nt < 4; ++nt)
#pragma unroll
        for (int r = 0; r < 8; ++r) o[nt].c[r] = 0.0f;

    unsigned short* pw = ldsP + w * 512;               // wave-private 16x32 tile
    const unsigned int* pr = (const unsigned int*)pw;

    for (int kc = 0; kc <= q0 + 15; kc += 32) {
        // ---- S = Q K^T for 32 key positions (two 16-col C fragments) ----
        CF s[2];
#pragma unroll
        for (int nt = 0; nt < 2; ++nt) {
            const unsigned int* kp =
                (const unsigned int*)(Kp + (size_t)(kc + nt * 16 + ln) * D_);
            AF b0, b1;
#pragma unroll
            for (int j = 0; j < 8; ++j) {
                const int p = pair_base(j, hi);
                b0.d[j] = kp[p];
                b1.d[j] = kp[16 + p];
            }
            CF sc;
#pragma unroll
            for (int r = 0; r < 8; ++r) sc.c[r] = 0.0f;
            sc.v = __builtin_amdgcn_wmma_f32_16x16x32_bf16(
                false, aq[0].v, false, b0.v, (short)0, sc.v, false, false);
            sc.v = __builtin_amdgcn_wmma_f32_16x16x32_bf16(
                false, aq[1].v, false, b1.v, (short)0, sc.v, false, false);
            s[nt] = sc;
        }

        // ---- scale (1/sqrt(64)=0.125) + causal mask in C/D layout ----
#pragma unroll
        for (int nt = 0; nt < 2; ++nt) {
            const int col = kc + nt * 16 + ln;
#pragma unroll
            for (int r = 0; r < 8; ++r) {
                const int row = q0 + r + hi * 8;
                const float x = s[nt].c[r] * 0.125f;
                s[nt].c[r] = (col <= row) ? x : -1e30f;
            }
        }

        // ---- online softmax (row reductions across 16-lane halves) ----
        float mnew[8];
#pragma unroll
        for (int r = 0; r < 8; ++r) {
            float v = fmaxf(s[0].c[r], s[1].c[r]);
#pragma unroll
            for (int msk = 1; msk < 16; msk <<= 1)
                v = fmaxf(v, __shfl_xor(v, msk, 32));
            mnew[r] = v;
        }
#pragma unroll
        for (int r = 0; r < 8; ++r) {
            const float mn  = fmaxf(mrun[r], mnew[r]);
            const float fac = __expf(mrun[r] - mn);
            mrun[r] = mn;
            const float e0 = __expf(s[0].c[r] - mn);
            const float e1 = __expf(s[1].c[r] - mn);
            s[0].c[r] = e0; s[1].c[r] = e1;
            float rs = e0 + e1;
#pragma unroll
            for (int msk = 1; msk < 16; msk <<= 1)
                rs += __shfl_xor(rs, msk, 32);
            lsum[r] = lsum[r] * fac + rs;
#pragma unroll
            for (int nt = 0; nt < 4; ++nt) o[nt].c[r] *= fac;
        }

        // ---- P: C-fragment layout -> LDS -> A-fragment layout ----
#pragma unroll
        for (int nt = 0; nt < 2; ++nt)
#pragma unroll
            for (int r = 0; r < 8; ++r)
                pw[(r + hi * 8) * 32 + nt * 16 + ln] = f2bf(s[nt].c[r]);
        __asm__ volatile("s_wait_dscnt 0" ::: "memory");
        AF apf;
#pragma unroll
        for (int j = 0; j < 8; ++j)
            apf.d[j] = pr[ln * 16 + pair_base(j, hi)];

        // ---- O += P @ V  (V stored transposed -> consecutive-pair loads) ----
#pragma unroll
        for (int nt = 0; nt < 4; ++nt) {
            const unsigned int* vp =
                (const unsigned int*)(Vt + (size_t)(nt * 16 + ln) * T_) + (kc >> 1);
            AF bv;
#pragma unroll
            for (int j = 0; j < 8; ++j) bv.d[j] = vp[pair_base(j, hi)];
            o[nt].v = __builtin_amdgcn_wmma_f32_16x16x32_bf16(
                false, apf.v, false, bv.v, (short)0, o[nt].v, false, false);
        }
    }

    // ---- normalize and store into [B,T,C] (head-interleaved) as bf16 ----
#pragma unroll
    for (int r = 0; r < 8; ++r) {
        const int   trow = q0 + r + hi * 8;
        const float inv  = 1.0f / lsum[r];
        unsigned short* dst = ao + ((size_t)(bb * T_ + trow)) * C_ + hh * D_;
#pragma unroll
        for (int nt = 0; nt < 4; ++nt)
            dst[nt * 16 + ln] = f2bf(o[nt].c[r] * inv);
    }
}

// ---------------------------------------------------------------------------
// Launch
// ---------------------------------------------------------------------------
extern "C" void kernel_launch(void* const* d_in, const int* in_sizes, int n_in,
                              void* d_out, int out_size, void* d_ws, size_t ws_size,
                              hipStream_t stream) {
    const float* x      = (const float*)d_in[0];
    const float* w_attn = (const float*)d_in[1];
    const float* b_attn = (const float*)d_in[2];
    const float* w_proj = (const float*)d_in[3];
    const float* b_proj = (const float*)d_in[4];
    float* out = (float*)d_out;

    char* ws = (char*)d_ws;
    size_t off = 0;
    auto carve = [&](size_t bytes) -> void* {
        void* p = ws + off;
        off += (bytes + 255) & ~(size_t)255;
        return p;
    };
    unsigned short* xb  = (unsigned short*)carve((size_t)M_ * C_ * 2);        // x bf16
    unsigned short* wat = (unsigned short*)carve((size_t)3 * C_ * C_ * 2);    // W_attn^T
    unsigned short* wpt = (unsigned short*)carve((size_t)C_ * C_ * 2);        // W_proj^T
    unsigned short* qb  = (unsigned short*)carve((size_t)M_ * C_ * 2);        // [B,H,T,D]
    unsigned short* kb  = (unsigned short*)carve((size_t)M_ * C_ * 2);        // [B,H,T,D]
    unsigned short* vtb = (unsigned short*)carve((size_t)M_ * C_ * 2);        // [B,H,D,T]
    unsigned short* ao  = (unsigned short*)carve((size_t)M_ * C_ * 2);        // [B,T,C]

    // 1) conversions / weight transposes
    {
        int n = M_ * C_;
        csa_cvt_f32_bf16<<<(n + 255) / 256, 256, 0, stream>>>(x, xb, n);
    }
    {
        int n = 3 * C_ * C_;
        csa_tr_f32_bf16<<<(n + 255) / 256, 256, 0, stream>>>(w_attn, wat, C_, 3 * C_);
    }
    {
        int n = C_ * C_;
        csa_tr_f32_bf16<<<(n + 255) / 256, 256, 0, stream>>>(w_proj, wpt, C_, C_);
    }

    // 2) QKV GEMM: [8192,768] x [768,2304]
    {
        dim3 grid(M_ / 128, (3 * C_) / 128);
        csa_gemm_wmma<<<grid, 256, 0, stream>>>(xb, wat, b_attn, C_, 0,
                                                qb, kb, vtb, nullptr);
    }

    // 3) causal flash attention
    {
        dim3 grid(B_ * H_, T_ / 128);
        csa_attn_wmma<<<grid, 256, 0, stream>>>(qb, kb, vtb, ao);
    }

    // 4) output projection: [8192,768] x [768,768] + bias -> f32 d_out
    {
        dim3 grid(M_ / 128, C_ / 128);
        csa_gemm_wmma<<<grid, 256, 0, stream>>>(ao, wpt, b_proj, C_, 1,
                                                nullptr, nullptr, nullptr, out);
    }
}
